// Self_Attention_120259084290
// MI455X (gfx1250) — compile-verified, matrix operand order checked by reading, not run on hardware
//
#include <hip/hip_runtime.h>

typedef __attribute__((ext_vector_type(16))) __bf16 v16bf;
typedef __attribute__((ext_vector_type(8)))  float  v8f;

#define NTOK 4096   // N = H*W
#define CCH  256    // channels
#define CQD  32     // q/k dim
#define VROW 40     // LDS V-tile row stride in shorts (32 data + 8 pad = 80B)

union U16 {
    v16bf v;
    float4 f[2];
    __bf16 b[16];
    unsigned short u[16];
};

// Hardware f32 -> bf16 (backend emits native v_cvt_pk_bf16_f32 where possible).
__device__ __forceinline__ unsigned short bf_bits(float x) {
    union { __bf16 h; unsigned short u; } cv;
    cv.h = (__bf16)x;
    return cv.u;
}

__device__ __forceinline__ v8f wmma_bf16(const v16bf& a, const v16bf& b, v8f c) {
    return __builtin_amdgcn_wmma_f32_16x16x32_bf16(false, a, false, b, (short)0, c,
                                                   false, false);
}

// Async global -> LDS copy (8 bytes per lane), tracked by ASYNCcnt.
__device__ __forceinline__ void async_load_b64(unsigned lds_off, const void* gaddr) {
    asm volatile("global_load_async_to_lds_b64 %0, %1, off"
                 :: "v"(lds_off), "v"(gaddr) : "memory");
}
__device__ __forceinline__ void wait_async0() {
    asm volatile("s_wait_asynccnt 0x0" ::: "memory");
}

// ---------------------------------------------------------------------------
// Stage 1: 1x1-conv projections.  Out = W (O x 256) * X (256 x N) + bias.
// Q,K stored transposed as bf16 [b][n][32]; V stored bf16 [b][256][n].
// One block: batch b, 64 columns of N.  256 threads = 8 waves.
// ---------------------------------------------------------------------------
__global__ __launch_bounds__(256)
void proj_kernel(const float* __restrict__ src,
                 const float* __restrict__ Wq, const float* __restrict__ bq,
                 const float* __restrict__ Wk, const float* __restrict__ bk,
                 const float* __restrict__ Wv, const float* __restrict__ bv,
                 unsigned short* __restrict__ Qt,
                 unsigned short* __restrict__ Kt,
                 unsigned short* __restrict__ Vc)
{
    __shared__ unsigned short Xs[64 * 256];   // [n][c] bf16, 32 KB

    const int b   = blockIdx.y;
    const int n0  = blockIdx.x * 64;
    const int tid = threadIdx.x;

    // Cooperative load of X tile: transpose (c-major -> n-major) + f32->bf16.
    const int nl = tid & 63;
    for (int c = tid >> 6; c < CCH; c += 4) {
        float x = src[((size_t)b * CCH + c) * NTOK + n0 + nl];
        Xs[nl * CCH + c] = bf_bits(x);
    }
    __syncthreads();

    const int wave = tid >> 5;
    const int lane = tid & 31;
    const int lm   = lane & 15;
    const int half = lane >> 4;

    // 320 output rows (32 Q + 32 K + 256 V) -> 20 row-tiles of 16, 8 waves loop.
    for (int ot = wave; ot < 20; ot += 8) {
        const int obase = ot * 16;
        const float* W;  const float* bias;  int orow;  int kind;
        if (obase < 32)      { W = Wq; bias = bq; orow = obase;      kind = 0; }
        else if (obase < 64) { W = Wk; bias = bk; orow = obase - 32; kind = 1; }
        else                 { W = Wv; bias = bv; orow = obase - 64; kind = 2; }

        v8f acc[4];
        #pragma unroll
        for (int t = 0; t < 4; ++t) acc[t] = (v8f){0.f,0.f,0.f,0.f,0.f,0.f,0.f,0.f};

        // Partial unroll: 1 iteration of load-ahead, bounded register pressure.
        #pragma unroll 2
        for (int kk = 0; kk < 8; ++kk) {            // K loop over C in steps of 32
            const int cb = kk * 32;
            // A operand: 16-bit A 16x32 layout — lane (M=lm) holds
            // K = {0..7,16..23} (half=0) or {8..15,24..31} (half=1), 2 per VGPR.
            const float* wr = W + (size_t)(orow + lm) * CCH + cb;
            U16 A;
            #pragma unroll
            for (int r = 0; r < 8; ++r) {
                int k0 = ((r < 4) ? 0 : 16) + half * 8 + (r & 3) * 2;
                A.b[2 * r + 0] = (__bf16)wr[k0 + 0];
                A.b[2 * r + 1] = (__bf16)wr[k0 + 1];
            }
            #pragma unroll
            for (int t = 0; t < 4; ++t) {
                // B operand (32x16): lane = column n, same K half-split.
                const int nrow = t * 16 + lm;
                U16 B;
                B.f[0] = *reinterpret_cast<const float4*>(&Xs[nrow * CCH + cb + half * 8]);
                B.f[1] = *reinterpret_cast<const float4*>(&Xs[nrow * CCH + cb + 16 + half * 8]);
                acc[t] = wmma_bf16(A.v, B.v, acc[t]);
            }
        }

        // D layout: vgpr r -> row M = r + 8*half, lane%16 -> col.
        #pragma unroll
        for (int r = 0; r < 8; ++r) {
            const int oo = orow + r + 8 * half;
            const float bb = bias[oo];
            #pragma unroll
            for (int t = 0; t < 4; ++t) {
                const int nn = n0 + t * 16 + lm;
                unsigned short h = bf_bits(acc[t][r] + bb);
                if (kind == 0)      Qt[((size_t)b * NTOK + nn) * CQD + oo] = h;
                else if (kind == 1) Kt[((size_t)b * NTOK + nn) * CQD + oo] = h;
                else                Vc[((size_t)b * CCH + oo) * NTOK + nn] = h;
            }
        }
    }
}

// ---------------------------------------------------------------------------
// Stage 2: fused flash attention + gamma*out + source.
// 256 threads = 8 waves: 4 i-tiles x 2 c-halves.  V tile staged into LDS via
// GLOBAL_LOAD_ASYNC_TO_LDS (no VGPR round-trip); the async transfer overlaps
// the S^T WMMAs + online softmax, then s_wait_asynccnt 0 precedes the
// barrier so the barrier implies the whole tile is visible.
// S^T = K_tile x Q so the softmax'd tile packs directly into the A operand
// of P x V (free transpose).
// ---------------------------------------------------------------------------
__global__ __launch_bounds__(256)
void flash_kernel(const unsigned short* __restrict__ Qt,   // [b][n][32]
                  const unsigned short* __restrict__ Kt,   // [b][n][32]
                  const unsigned short* __restrict__ Vc,   // [b][256][n]
                  const float* __restrict__ src,
                  const float* __restrict__ gamma,
                  float* __restrict__ out)
{
    __shared__ unsigned short Vs[CCH * VROW];   // 256 rows x 80B = 20 KB

    const int tid   = threadIdx.x;
    const int b     = blockIdx.y;
    const int wave  = tid >> 5;
    const int lane  = tid & 31;
    const int lm    = lane & 15;
    const int half  = lane >> 4;
    const int i0    = (blockIdx.x * 4 + (wave >> 1)) * 16;
    const int chalf = (wave & 1) * 128;           // this wave's 128 channels

    // Cooperative V-tile fill mapping: 8 threads per row, 8B per thread.
    const int crow = tid >> 3;        // 0..31 (x8 passes = 256 rows)
    const int jo8  = (tid & 7) * 4;   // short offset within row

    // Per-lane LDS byte address for the async fill (addr[31:0] = LDS offset).
    const unsigned vs_base = (unsigned)(uintptr_t)(void*)&Vs[0];

    // Q as B operand (32 x 16, d x i): lane = i, two 16B loads select d-half.
    U16 Qb;
    {
        const float4* qp = reinterpret_cast<const float4*>(
            Qt + ((size_t)b * NTOK + i0 + lm) * CQD);
        Qb.f[0] = qp[half];        // d = half*8 .. half*8+7
        Qb.f[1] = qp[half + 2];    // d = 16+half*8 ..
    }

    v8f O[8];
    #pragma unroll
    for (int ct = 0; ct < 8; ++ct) O[ct] = (v8f){0.f,0.f,0.f,0.f,0.f,0.f,0.f,0.f};
    float m = -3.0e38f;   // running row max (keyed to i = lm)
    float l = 0.0f;       // running partial row sum (this lane's 8-of-16 share)

    const float g = gamma[0];

    for (int jt = 0; jt < NTOK; jt += 32) {
        // ---- async-stage V tile (256 x 32 bf16) into LDS, 64B/row coalesced
        __syncthreads();   // previous iteration's readers are done
        #pragma unroll
        for (int rr = 0; rr < 8; ++rr) {
            const int c = rr * 32 + crow;
            async_load_b64(vs_base + (unsigned)(c * VROW + jo8) * 2,
                           &Vc[((size_t)b * CCH + c) * NTOK + jt + jo8]);
        }

        // K tiles as A operands for S^T = K x Q (two 16-row subtiles),
        // loaded while the async LDS fill is in flight.
        U16 A0, A1;
        {
            const float4* k0 = reinterpret_cast<const float4*>(
                Kt + ((size_t)b * NTOK + jt + lm) * CQD);
            A0.f[0] = k0[half];  A0.f[1] = k0[half + 2];
            const float4* k1 = reinterpret_cast<const float4*>(
                Kt + ((size_t)b * NTOK + jt + 16 + lm) * CQD);
            A1.f[0] = k1[half];  A1.f[1] = k1[half + 2];
        }
        if (jt + 32 < NTOK)
            __builtin_prefetch(Kt + ((size_t)b * NTOK + jt + 32 + lm) * CQD, 0, 3);

        v8f z = (v8f){0.f,0.f,0.f,0.f,0.f,0.f,0.f,0.f};
        v8f S0 = wmma_bf16(A0.v, Qb.v, z);   // S^T: vgpr r -> j, lane%16 -> i
        v8f S1 = wmma_bf16(A1.v, Qb.v, z);

        // --- online softmax (per row i = lm; halves merged via shfl_xor 16) ---
        float tmax = S0[0];
        #pragma unroll
        for (int r = 1; r < 8; ++r) tmax = fmaxf(tmax, S0[r]);
        #pragma unroll
        for (int r = 0; r < 8; ++r) tmax = fmaxf(tmax, S1[r]);
        tmax = fmaxf(tmax, __shfl_xor(tmax, 16));
        const float nm = fmaxf(m, tmax);
        const float sc = __expf(m - nm);
        m = nm;
        l *= sc;

        float p[16];
        #pragma unroll
        for (int r = 0; r < 8; ++r) {
            p[r]     = __expf(S0[r] - m);
            p[8 + r] = __expf(S1[r] - m);
            l += p[r] + p[8 + r];
        }

        // Pack P directly into A-operand layout (free transpose from S^T).
        U16 Pa;
        #pragma unroll
        for (int e = 0; e < 16; ++e) Pa.b[e] = (__bf16)p[e];

        // Rescale O: O rows are i = r + 8*half; sc is keyed to i = lm.
        float sM[8];
        #pragma unroll
        for (int r = 0; r < 8; ++r) sM[r] = __shfl(sc, r + 8 * half);
        #pragma unroll
        for (int ct = 0; ct < 8; ++ct) {
            #pragma unroll
            for (int r = 0; r < 8; ++r) O[ct][r] *= sM[r];
        }

        wait_async0();     // this wave's async portion has landed in LDS
        __syncthreads();   // => entire V tile visible to all waves

        // P (16i x 32j) x V^T (32j x 16c) accumulate over this wave's c-half,
        // V operand read from LDS (low latency -> little load batching).
        #pragma unroll
        for (int ct = 0; ct < 8; ++ct) {
            const int c = chalf + ct * 16 + lm;
            U16 Vb;
            Vb.f[0] = *reinterpret_cast<const float4*>(&Vs[c * VROW + half * 8]);
            Vb.f[1] = *reinterpret_cast<const float4*>(&Vs[c * VROW + 16 + half * 8]);
            O[ct] = wmma_bf16(Pa.v, Vb.v, O[ct]);
        }
    }

    // Epilogue: normalize, gamma-scale, residual add.
    const float lt  = l + __shfl_xor(l, 16);
    const float rcp = 1.0f / lt;              // keyed to i = lm
    float rM[8];
    #pragma unroll
    for (int r = 0; r < 8; ++r) rM[r] = __shfl(rcp, r + 8 * half);

    #pragma unroll
    for (int ct = 0; ct < 8; ++ct) {
        const int c = chalf + ct * 16 + lm;
        #pragma unroll
        for (int r = 0; r < 8; ++r) {
            const int i = i0 + r + 8 * half;
            const size_t idx = ((size_t)b * CCH + c) * NTOK + i;
            out[idx] = g * (O[ct][r] * rM[r]) + src[idx];
        }
    }
}

// ---------------------------------------------------------------------------
extern "C" void kernel_launch(void* const* d_in, const int* in_sizes, int n_in,
                              void* d_out, int out_size, void* d_ws, size_t ws_size,
                              hipStream_t stream)
{
    (void)in_sizes; (void)n_in; (void)out_size; (void)ws_size;
    const float* src   = (const float*)d_in[0];
    const float* Wq    = (const float*)d_in[1];
    const float* bq    = (const float*)d_in[2];
    const float* Wk    = (const float*)d_in[3];
    const float* bk    = (const float*)d_in[4];
    const float* Wv    = (const float*)d_in[5];
    const float* bv    = (const float*)d_in[6];
    const float* gamma = (const float*)d_in[7];
    float* out = (float*)d_out;

    // Workspace: Qt (2MB) | Kt (2MB) | Vc (16MB), all bf16.
    unsigned short* Qt = (unsigned short*)d_ws;
    unsigned short* Kt = Qt + (size_t)8 * NTOK * CQD;
    unsigned short* Vc = Kt + (size_t)8 * NTOK * CQD;

    hipLaunchKernelGGL(proj_kernel, dim3(NTOK / 64, 8), dim3(256), 0, stream,
                       src, Wq, bq, Wk, bk, Wv, bv, Qt, Kt, Vc);
    hipLaunchKernelGGL(flash_kernel, dim3(NTOK / 64, 8), dim3(256), 0, stream,
                       Qt, Kt, Vc, src, gamma, out);
}
